// LSTMclassifier_67912022885084
// MI455X (gfx1250) — compile-verified, matrix operand order checked by reading, not run on hardware
//

#include <hip/hip_runtime.h>
#include <hip/hip_bf16.h>
#include <math.h>

// ---------------- CDNA5 vector types ----------------
typedef __attribute__((ext_vector_type(16))) _Float16 v16h;
typedef __attribute__((ext_vector_type(8)))  _Float16 v8h;
typedef __attribute__((ext_vector_type(8)))  float    v8f;

#define BATCH 16384
#define TSTEPS 10
#define INDIM 256
#define HDIM 16
#define GDIM 64            // 4*H
#define ROWS (BATCH * TSTEPS)   // 163840
#define FEAT (TSTEPS * HDIM)    // 160

__device__ __forceinline__ float sigmoidf_(float x) {
    return 1.0f / (1.0f + __expf(-x));
}

// ---------------- kernel 0: prep (f32->f16 weights, zero BN accumulators) ---
__global__ void k_prep(const float* __restrict__ w_ih0,
                       _Float16* __restrict__ wh,
                       float* __restrict__ sums /* [320] */) {
    int i = blockIdx.x * blockDim.x + threadIdx.x;
    if (i < GDIM * INDIM) wh[i] = (_Float16)w_ih0[i];
    if (i < 2 * FEAT) sums[i] = 0.0f;
}

// ---------------- kernel 1: WMMA GEMM  Y[r,g] = sum_k X[r,k]*W[g,k] --------
// A = W tile (M=16 gates x K=32, f16), B = X tile (K=32 x N=16 rows, f16).
// Each wave: 16 rows x 64 gates, 8 K-iterations -> 32 v_wmma per wave.
__global__ void k_gemm_xw0(const float* __restrict__ X,
                           const _Float16* __restrict__ Wh,
                           float* __restrict__ Y) {
    const int lane  = threadIdx.x & 31;
    const int wave  = threadIdx.x >> 5;
    const int nrow  = lane & 15;     // row-in-tile (B/N) or gate-in-tile (A/M)
    const int khalf = lane >> 4;     // lo/hi half of wave -> K sub-block
    const long r    = (long)(blockIdx.x * 8 + wave) * 16 + nrow;  // < ROWS

    const float* __restrict__ xrow = X + r * INDIM;

    v8f acc[4] = {};

    #pragma unroll
    for (int ki = 0; ki < 8; ++ki) {
        const int kb = ki * 32 + khalf * 8;   // K = kb+{0..7}, 16+kb+{0..7}

        // ---- B fragment: X row r, converted f32 -> f16 (layout §7.12.2) ----
        const float4 q0 = *(const float4*)(xrow + kb);
        const float4 q1 = *(const float4*)(xrow + kb + 4);
        const float4 q2 = *(const float4*)(xrow + kb + 16);
        const float4 q3 = *(const float4*)(xrow + kb + 20);
        v16h b;
        b[0]  = (_Float16)q0.x; b[1]  = (_Float16)q0.y;
        b[2]  = (_Float16)q0.z; b[3]  = (_Float16)q0.w;
        b[4]  = (_Float16)q1.x; b[5]  = (_Float16)q1.y;
        b[6]  = (_Float16)q1.z; b[7]  = (_Float16)q1.w;
        b[8]  = (_Float16)q2.x; b[9]  = (_Float16)q2.y;
        b[10] = (_Float16)q2.z; b[11] = (_Float16)q2.w;
        b[12] = (_Float16)q3.x; b[13] = (_Float16)q3.y;
        b[14] = (_Float16)q3.z; b[15] = (_Float16)q3.w;

        // ---- 4 A fragments (gate tiles), pre-converted f16 weights ----
        #pragma unroll
        for (int mt = 0; mt < 4; ++mt) {
            const int g = mt * 16 + nrow;
            const _Float16* __restrict__ wrow = Wh + g * INDIM + kb;
            const v8h alo = *(const v8h*)(wrow);
            const v8h ahi = *(const v8h*)(wrow + 16);
            v16h a;
            #pragma unroll
            for (int e = 0; e < 8; ++e) { a[e] = alo[e]; a[e + 8] = ahi[e]; }

            acc[mt] = __builtin_amdgcn_wmma_f32_16x16x32_f16(
                false, a, false, b, (short)0, acc[mt], false, false);
        }
    }

    // ---- store: lane holds Y[r, mt*16 + khalf*8 + v], v=0..7 (contiguous) --
    float* __restrict__ yrow = Y + r * GDIM;
    #pragma unroll
    for (int mt = 0; mt < 4; ++mt) {
        float4 s0, s1;
        s0.x = acc[mt][0]; s0.y = acc[mt][1]; s0.z = acc[mt][2]; s0.w = acc[mt][3];
        s1.x = acc[mt][4]; s1.y = acc[mt][5]; s1.z = acc[mt][6]; s1.w = acc[mt][7];
        *(float4*)(yrow + mt * 16 + khalf * 8)     = s0;
        *(float4*)(yrow + mt * 16 + khalf * 8 + 4) = s1;
    }
}

// ---------------- kernel 2: fused 2-layer LSTM recurrence (fp32) -----------
__global__ void k_lstm(const float* __restrict__ Y0,
                       const float* __restrict__ whh0,
                       const float* __restrict__ wih1,
                       const float* __restrict__ whh1,
                       const float* __restrict__ bih0, const float* __restrict__ bhh0,
                       const float* __restrict__ bih1, const float* __restrict__ bhh1,
                       float* __restrict__ flatOut) {
    __shared__ float sWhh0[GDIM * HDIM];
    __shared__ float sWih1[GDIM * HDIM];
    __shared__ float sWhh1[GDIM * HDIM];
    __shared__ float sB0[GDIM];
    __shared__ float sB1[GDIM];

    for (int i = threadIdx.x; i < GDIM * HDIM; i += blockDim.x) {
        sWhh0[i] = whh0[i];
        sWih1[i] = wih1[i];
        sWhh1[i] = whh1[i];
    }
    for (int i = threadIdx.x; i < GDIM; i += blockDim.x) {
        sB0[i] = bih0[i] + bhh0[i];
        sB1[i] = bih1[i] + bhh1[i];
    }
    __syncthreads();

    const int s = blockIdx.x * blockDim.x + threadIdx.x;  // sample
    float h0[HDIM], c0[HDIM], h1[HDIM], c1[HDIM], hn[HDIM];
    #pragma unroll
    for (int j = 0; j < HDIM; ++j) { h0[j] = 0.f; c0[j] = 0.f; h1[j] = 0.f; c1[j] = 0.f; }

    const float* __restrict__ xg = Y0 + (long)s * TSTEPS * GDIM;
    float* __restrict__ orow = flatOut + (long)s * FEAT;

    for (int t = 0; t < TSTEPS; ++t) {
        const float* __restrict__ xt = xg + t * GDIM;
        // ---- layer 0 : gates = xg + b + h0 @ Whh0^T ----
        #pragma unroll
        for (int j = 0; j < HDIM; ++j) {
            float gi = xt[j]      + sB0[j];
            float gf = xt[j + 16] + sB0[j + 16];
            float gg = xt[j + 32] + sB0[j + 32];
            float go = xt[j + 48] + sB0[j + 48];
            #pragma unroll
            for (int k = 0; k < HDIM; ++k) {
                const float hk = h0[k];
                gi = fmaf(sWhh0[j * 16 + k],        hk, gi);
                gf = fmaf(sWhh0[(j + 16) * 16 + k], hk, gf);
                gg = fmaf(sWhh0[(j + 32) * 16 + k], hk, gg);
                go = fmaf(sWhh0[(j + 48) * 16 + k], hk, go);
            }
            const float cc = sigmoidf_(gf) * c0[j] + sigmoidf_(gi) * tanhf(gg);
            c0[j] = cc;
            hn[j] = sigmoidf_(go) * tanhf(cc);
        }
        #pragma unroll
        for (int j = 0; j < HDIM; ++j) h0[j] = hn[j];

        // ---- layer 1 : gates = b + h0 @ Wih1^T + h1 @ Whh1^T ----
        #pragma unroll
        for (int j = 0; j < HDIM; ++j) {
            float gi = sB1[j], gf = sB1[j + 16], gg = sB1[j + 32], go = sB1[j + 48];
            #pragma unroll
            for (int k = 0; k < HDIM; ++k) {
                const float xk = h0[k], hk = h1[k];
                gi = fmaf(sWih1[j * 16 + k],        xk, fmaf(sWhh1[j * 16 + k],        hk, gi));
                gf = fmaf(sWih1[(j + 16) * 16 + k], xk, fmaf(sWhh1[(j + 16) * 16 + k], hk, gf));
                gg = fmaf(sWih1[(j + 32) * 16 + k], xk, fmaf(sWhh1[(j + 32) * 16 + k], hk, gg));
                go = fmaf(sWih1[(j + 48) * 16 + k], xk, fmaf(sWhh1[(j + 48) * 16 + k], hk, go));
            }
            const float cc = sigmoidf_(gf) * c1[j] + sigmoidf_(gi) * tanhf(gg);
            c1[j] = cc;
            hn[j] = sigmoidf_(go) * tanhf(cc);
        }
        #pragma unroll
        for (int j = 0; j < HDIM; ++j) {
            h1[j] = hn[j];
            orow[t * 16 + j] = hn[j];
        }
    }
}

// ---------------- kernel 3: BN batch statistics ----------------------------
__global__ void k_bnstats(const float* __restrict__ flat,
                          float* __restrict__ sums /* [320] */) {
    const int f  = threadIdx.x;          // 160 threads
    const int r0 = blockIdx.x * 128;     // 128 blocks x 128 rows = 16384
    float s = 0.f, ss = 0.f;
    for (int i = 0; i < 128; ++i) {
        const float v = flat[(long)(r0 + i) * FEAT + f];
        s += v;
        ss = fmaf(v, v, ss);
    }
    atomicAdd(&sums[f], s);
    atomicAdd(&sums[FEAT + f], ss);
}

__global__ void k_bnfinalize(const float* __restrict__ sums,
                             const float* __restrict__ gamma,
                             const float* __restrict__ beta,
                             float* __restrict__ scsh /* [320] */) {
    const int f = threadIdx.x;           // 160 threads
    const float inv_n = 1.0f / (float)BATCH;
    const float mean = sums[f] * inv_n;
    const float var  = sums[FEAT + f] * inv_n - mean * mean;  // biased
    const float sc   = gamma[f] * rsqrtf(var + 1e-5f);
    scsh[f]        = sc;
    scsh[FEAT + f] = beta[f] - mean * sc;
}

// ---------------- kernel 4: BN-apply + LeakyReLU + FC + softmax ------------
__global__ void k_head(const float* __restrict__ flat,
                       const float* __restrict__ scsh,
                       const float* __restrict__ ag,
                       const float* __restrict__ fcw,   // [2,162]
                       const float* __restrict__ fcb,   // [2]
                       float* __restrict__ out) {
    const int lane = threadIdx.x & 31;
    const int s    = (blockIdx.x * blockDim.x + threadIdx.x) >> 5;  // sample/wave
    const float* __restrict__ row = flat + (long)s * FEAT;

    float a0 = 0.f, a1 = 0.f;
    #pragma unroll
    for (int i = 0; i < 5; ++i) {
        const int f = lane * 5 + i;      // 32 lanes x 5 = 160 features
        float v = fmaf(row[f], scsh[f], scsh[FEAT + f]);
        v = v >= 0.f ? v : 0.01f * v;    // LeakyReLU
        a0 = fmaf(v, fcw[f],       a0);
        a1 = fmaf(v, fcw[162 + f], a1);
    }
    #pragma unroll
    for (int off = 16; off > 0; off >>= 1) {
        a0 += __shfl_xor(a0, off, 32);
        a1 += __shfl_xor(a1, off, 32);
    }
    if (lane == 0) {
        const float g0 = ag[s * 2], g1 = ag[s * 2 + 1];
        const float l0 = a0 + fcw[160]       * g0 + fcw[161]       * g1 + fcb[0];
        const float l1 = a1 + fcw[162 + 160] * g0 + fcw[162 + 161] * g1 + fcb[1];
        const float m  = fmaxf(l0, l1);
        const float e0 = __expf(l0 - m), e1 = __expf(l1 - m);
        const float inv = 1.0f / (e0 + e1);
        out[s * 2]     = e0 * inv;
        out[s * 2 + 1] = e1 * inv;
    }
}

// ---------------- launcher --------------------------------------------------
extern "C" void kernel_launch(void* const* d_in, const int* in_sizes, int n_in,
                              void* d_out, int out_size, void* d_ws, size_t ws_size,
                              hipStream_t stream) {
    const float* inputs = (const float*)d_in[0];   // [16384,10,256]
    const float* ageg   = (const float*)d_in[1];   // [16384,2]
    const float* w_ih0  = (const float*)d_in[2];   // [64,256]
    const float* w_hh0  = (const float*)d_in[3];   // [64,16]
    const float* b_ih0  = (const float*)d_in[4];
    const float* b_hh0  = (const float*)d_in[5];
    const float* w_ih1  = (const float*)d_in[6];   // [64,16]
    const float* w_hh1  = (const float*)d_in[7];   // [64,16]
    const float* b_ih1  = (const float*)d_in[8];
    const float* b_hh1  = (const float*)d_in[9];
    const float* gamma  = (const float*)d_in[10];  // [160]
    const float* beta   = (const float*)d_in[11];  // [160]
    const float* fc_w   = (const float*)d_in[12];  // [2,162]
    const float* fc_b   = (const float*)d_in[13];  // [2]
    float* out = (float*)d_out;                    // [16384,2]

    // workspace layout (256B-aligned regions)
    char* base = (char*)d_ws;
    float*    Y0   = (float*)(base);                                    // 41,943,040 B
    float*    flat = (float*)(base + 41943040);                         // 10,485,760 B
    _Float16* wh   = (_Float16*)(base + 41943040 + 10485760);           //     32,768 B
    float*    sums = (float*)(base + 41943040 + 10485760 + 32768);      //      1,280 B
    float*    scsh = (float*)(base + 41943040 + 10485760 + 32768 + 1280);

    // 0) convert W_ih0 -> f16, zero BN accumulators (every call; deterministic)
    k_prep<<<64, 256, 0, stream>>>(w_ih0, wh, sums);

    // 1) dominant GEMM via v_wmma_f32_16x16x32_f16: 10240 waves, 8 waves/block
    k_gemm_xw0<<<1280, 256, 0, stream>>>(inputs, wh, Y0);

    // 2) fused 2-layer LSTM recurrence (fp32, weights in LDS)
    k_lstm<<<64, 256, 0, stream>>>(Y0, w_hh0, w_ih1, w_hh1,
                                   b_ih0, b_hh0, b_ih1, b_hh1, flat);

    // 3) BN batch stats + finalize
    k_bnstats<<<128, 160, 0, stream>>>(flat, sums);
    k_bnfinalize<<<1, 160, 0, stream>>>(sums, gamma, beta, scsh);

    // 4) head: BN apply + LeakyReLU + FC + softmax (one wave per sample)
    k_head<<<2048, 256, 0, stream>>>(flat, scsh, ageg, fc_w, fc_b, out);
}
